// NNMF2dGrouped_51247549776344
// MI455X (gfx1250) — matrix operand order repaired and checked by reading, not compiled, not run on hardware
//
#include <hip/hip_runtime.h>
#include <hip/hip_bf16.h>

// ---------------------------------------------------------------------------
// Grouped NMF (20 multiplicative-update iterations), MI455X / gfx1250.
//
// Per wave: 16 pixels of one (batch, group).  All 20 iterations fused; h and
// xr live in registers in the WMMA "transposed-D" layout so that
//   pass1: rec^T = A(w^T tiles, const) x B(h^T)      [2 WMMA k-steps x 4 tiles]
//   pass2: v^T   = A(w  tiles, const)  x B(q^T)      [2 WMMA k-steps x 4 tiles]
// D-layout -> next B-layout conversion needs only f32->bf16 packing, a
// lane-half tile select, and half-wave broadcasts (ds_bpermute) -- no LDS.
// 16 x v_wmma_f32_16x16x32_bf16 per iteration per wave.
// ---------------------------------------------------------------------------

typedef __attribute__((ext_vector_type(16))) __bf16 v16bf;
typedef __attribute__((ext_vector_type(8)))  float  v8f;

union BF16x16 {
  v16bf    v;
  unsigned u[8];
  uint4    u4[2];
};

__device__ __forceinline__ unsigned pk_bf16(float lo, float hi) {
  union { __bf16 b[2]; unsigned u; } t;
  t.b[0] = (__bf16)lo;
  t.b[1] = (__bf16)hi;
  return t.u;
}

__device__ __forceinline__ unsigned shflu(unsigned v, int src) {
  return (unsigned)__shfl((int)v, src, 32);
}

// T[jj][r] at lane l holds M[16*jj + r + 8*(l>=16)][l&15] (transposed-D layout).
// Produces B[s] (s = K-step) with: lane l element e == bf16(M[32s + 16*(l>=16) + e][l&15]),
// i.e. contiguous k order (k = 16*half + e), the same order the prepped A tiles use.
__device__ __forceinline__ void build_B(const float T[4][8], BF16x16 B[2],
                                        int ln, int hl) {
  unsigned P[4][4];
#pragma unroll
  for (int j = 0; j < 4; ++j)
#pragma unroll
    for (int d = 0; d < 4; ++d)
      P[j][d] = pk_bf16(T[j][2 * d], T[j][2 * d + 1]);

#pragma unroll
  for (int s = 0; s < 2; ++s) {
#pragma unroll
    for (int d = 0; d < 4; ++d) {
      unsigned lo0 = shflu(P[2 * s + 0][d], ln);       // rows 16*(2s)   + 2d(+1)
      unsigned lo1 = shflu(P[2 * s + 1][d], ln);       // rows 16*(2s+1) + 2d(+1)
      unsigned hi0 = shflu(P[2 * s + 0][d], ln + 16);  // +8 rows
      unsigned hi1 = shflu(P[2 * s + 1][d], ln + 16);
      B[s].u[d]     = hl ? lo1 : lo0;
      B[s].u[d + 4] = hl ? hi1 : hi0;
    }
  }
}

// ---------------------------------------------------------------------------
// Prep kernel: wn = |w| / (sum_i |w| + eps), then pre-swizzle into the exact
// per-lane WMMA A-operand dwords (bf16) so the main kernel does coalesced
// b128 loads.  ws layout: [g][tile 0..15][lane 0..31][8 dwords];
// tiles 0..7 = w^T A-tiles (jj*2+s), tiles 8..15 = w A-tiles.
// ---------------------------------------------------------------------------
__global__ void nmf_wprep(const float* __restrict__ w, unsigned* __restrict__ ws) {
  __shared__ float wn[64][64];
  const int g = blockIdx.x;
  const int tid = threadIdx.x;  // 64 threads: one output row o each

  const float* row = w + (size_t)(g * 64 + tid) * 64;
  float a[64];
  float s = 0.f;
#pragma unroll
  for (int i = 0; i < 64; ++i) { a[i] = fabsf(row[i]); s += a[i]; }
  float inv = 1.f / (s + 1e-19f);
#pragma unroll
  for (int i = 0; i < 64; ++i) wn[tid][i] = a[i] * inv;
  __syncthreads();

  if (tid < 32) {
    const int ln = tid & 15, hl = tid >> 4;
#pragma unroll
    for (int jj = 0; jj < 4; ++jj)
#pragma unroll
      for (int s2 = 0; s2 < 2; ++s2) {
        unsigned d1[8], d2[8];
#pragma unroll
        for (int d = 0; d < 8; ++d) {
          int e0 = 2 * d, e1 = 2 * d + 1;
          int k0 = 32 * s2 + e0 + 16 * hl;  // A-layout K
          int k1 = 32 * s2 + e1 + 16 * hl;
          d1[d] = pk_bf16(wn[k0][16 * jj + ln], wn[k1][16 * jj + ln]);  // w^T
          d2[d] = pk_bf16(wn[16 * jj + ln][k0], wn[16 * jj + ln][k1]);  // w
        }
        unsigned* p1 = ws + (((size_t)(g * 16 + (jj * 2 + s2)) * 32 + tid) * 8);
        unsigned* p2 = ws + (((size_t)(g * 16 + 8 + jj * 2 + s2) * 32 + tid) * 8);
#pragma unroll
        for (int d = 0; d < 8; ++d) { p1[d] = d1[d]; p2[d] = d2[d]; }
      }
  }
}

// ---------------------------------------------------------------------------
// Main fused kernel.  4 waves / block; wave wid = (b*8+g)*64 + pixel_tile.
// ---------------------------------------------------------------------------
__global__ __launch_bounds__(128) void nmf_main(const float* __restrict__ x,
                                                const uint4* __restrict__ wsw,
                                                float* __restrict__ out) {
  const int lane = threadIdx.x & 31;
  const int wav  = threadIdx.x >> 5;
  const int wid  = blockIdx.x * 4 + wav;     // 0..16383
  const int t    = wid & 63;                 // pixel tile within image
  const int g    = (wid >> 6) & 7;
  const int b    = wid >> 9;
  const int ln   = lane & 15;
  const int hl   = lane >> 4;
  const int pix  = t * 16 + ln;

  // ---- constant weight A-operands (pre-swizzled), kept in VGPRs ----
  BF16x16 A1[4][2], A2[4][2];
#pragma unroll
  for (int jj = 0; jj < 4; ++jj)
#pragma unroll
    for (int s = 0; s < 2; ++s) {
      const uint4* p1 = wsw + (((size_t)(g * 16 + (jj * 2 + s)) * 32 + lane) * 2);
      const uint4* p2 = wsw + (((size_t)(g * 16 + 8 + jj * 2 + s) * 32 + lane) * 2);
      A1[jj][s].u4[0] = p1[0]; A1[jj][s].u4[1] = p1[1];
      A2[jj][s].u4[0] = p2[0]; A2[jj][s].u4[1] = p2[1];
    }

  // ---- load x tile in transposed-D layout; normalize over in-channels ----
  // XR[jj][r] = x[b][g*64 + i][pix], i = 16*jj + r + 8*hl
  const float* xb = x + (size_t)(b * 8 + g) * 64 * 1024;
  float XR[4][8];
  float xs = 0.f;
#pragma unroll
  for (int jj = 0; jj < 4; ++jj)
#pragma unroll
    for (int r = 0; r < 8; ++r) {
      float v = xb[(size_t)(16 * jj + r + 8 * hl) * 1024 + pix];
      XR[jj][r] = v;
      xs += v;
    }
  xs += __shfl_xor(xs, 16, 32);  // other lane-half holds the other 32 channels
  float xinv = __builtin_amdgcn_rcpf(xs + 1e-19f);
#pragma unroll
  for (int jj = 0; jj < 4; ++jj)
#pragma unroll
    for (int r = 0; r < 8; ++r) XR[jj][r] *= xinv;

  // ---- h state, same layout, init 1/64 ----
  float H[4][8];
#pragma unroll
  for (int jj = 0; jj < 4; ++jj)
#pragma unroll
    for (int r = 0; r < 8; ++r) H[jj][r] = 0.015625f;

  for (int it = 0; it < 20; ++it) {
    // pass 1: rec^T = w^T x h^T ; then q = xr / (rec + eps)
    BF16x16 Bh[2];
    build_B(H, Bh, ln, hl);
    float Q[4][8];
#pragma unroll
    for (int jj = 0; jj < 4; ++jj) {
      v8f acc = {};
      acc = __builtin_amdgcn_wmma_f32_16x16x32_bf16(
          false, A1[jj][0].v, false, Bh[0].v, (short)0, acc, false, false);
      acc = __builtin_amdgcn_wmma_f32_16x16x32_bf16(
          false, A1[jj][1].v, false, Bh[1].v, (short)0, acc, false, false);
#pragma unroll
      for (int r = 0; r < 8; ++r)
        Q[jj][r] = XR[jj][r] * __builtin_amdgcn_rcpf(acc[r] + 1e-20f);
    }

    // pass 2: v^T = w x q^T ; h *= v ; normalize over out-channels
    BF16x16 Bq[2];
    build_B(Q, Bq, ln, hl);
    float hs = 0.f;
#pragma unroll
    for (int jj = 0; jj < 4; ++jj) {
      v8f acc = {};
      acc = __builtin_amdgcn_wmma_f32_16x16x32_bf16(
          false, A2[jj][0].v, false, Bq[0].v, (short)0, acc, false, false);
      acc = __builtin_amdgcn_wmma_f32_16x16x32_bf16(
          false, A2[jj][1].v, false, Bq[1].v, (short)0, acc, false, false);
#pragma unroll
      for (int r = 0; r < 8; ++r) { H[jj][r] *= acc[r]; hs += H[jj][r]; }
    }
    hs += __shfl_xor(hs, 16, 32);
    float sc = __builtin_amdgcn_rcpf(hs + 1e-19f);
#pragma unroll
    for (int jj = 0; jj < 4; ++jj)
#pragma unroll
      for (int r = 0; r < 8; ++r) H[jj][r] *= sc;
  }

  // Final global normalization over all 512 channels: per-group sums are 1.0
  // to within ~1e-20 relative (eps=1e-19 vanishes in fp32), so divide by 8.
  float* ob = out + (size_t)(b * 8 + g) * 64 * 1024;
#pragma unroll
  for (int jj = 0; jj < 4; ++jj)
#pragma unroll
    for (int r = 0; r < 8; ++r)
      ob[(size_t)(16 * jj + r + 8 * hl) * 1024 + pix] = H[jj][r] * 0.125f;
}

extern "C" void kernel_launch(void* const* d_in, const int* in_sizes, int n_in,
                              void* d_out, int out_size, void* d_ws, size_t ws_size,
                              hipStream_t stream) {
  const float* x = (const float*)d_in[0];      // (32, 512, 32, 32) f32
  const float* w = (const float*)d_in[1];      // (8, 64, 64) f32
  float* out = (float*)d_out;                  // (32, 512, 32, 32) f32
  unsigned* wswz = (unsigned*)d_ws;            // 128 KB: swizzled bf16 A-tiles

  nmf_wprep<<<8, 64, 0, stream>>>(w, wswz);
  // 16384 waves (32 b x 8 g x 64 pixel-tiles), 4 waves per 128-thread block
  nmf_main<<<4096, 128, 0, stream>>>(x, (const uint4*)wswz, out);
}
